// CGCNN_75591424409904
// MI455X (gfx1250) — matrix operand order repaired and checked by reading, not compiled, not run on hardware
//
#include <hip/hip_runtime.h>

// ---------------------------------------------------------------------------
// CGCNN forward for MI455X (gfx1250).  Conv GEMMs run on v_wmma_f32_16x16x32_f16
// with f32 accumulation; BN statistics are computed by a first WMMA pass (keeps
// t in registers, reduces sum/sumsq), and a second WMMA pass applies BN + the
// sigmoid*softplus gate + neighbor sum fully in registers/LDS.  t (246MB/layer)
// is never written to memory.
// ---------------------------------------------------------------------------

typedef __attribute__((ext_vector_type(16))) _Float16 v16h;
typedef __attribute__((ext_vector_type(8)))  float    v8f;

#define NN 40000
#define MM 12
#define AA 64
#define BB 64
#define LL 3
#define CC 128
#define F0 92
#define HH 128
#define KDIM 192   // 2A + B
#define DDIM 128   // 2A
#define NM (NN * MM)

__device__ __forceinline__ float softplusf(float x) {
    return (x > 20.f) ? x : log1pf(__expf(x));
}
__device__ __forceinline__ float sigmoidf(float x) {
    return 1.f / (1.f + __expf(-x));
}

__device__ __forceinline__ v8f wmma_f16(v16h a, v16h b, v8f c) {
    return __builtin_amdgcn_wmma_f32_16x16x32_f16(false, a, false, b,
                                                  (short)0, c, false, false);
}

// Build a 16x32 f16 A-fragment for this lane from an f32 row pointer.
// Lane layout (ISA 7.12.2): lanes 0-15 M=row, K = kb+{0..7}, kb+16+{0..7},
// kb = 0 (lanes 0-15) or 8 (lanes 16-31).
__device__ __forceinline__ v16h load_afrag(const float* __restrict__ p, int kb) {
    float4 x0 = *(const float4*)(p + kb);
    float4 x1 = *(const float4*)(p + kb + 4);
    float4 x2 = *(const float4*)(p + kb + 16);
    float4 x3 = *(const float4*)(p + kb + 20);
    v16h a;
    a[0]  = (_Float16)x0.x;  a[1]  = (_Float16)x0.y;
    a[2]  = (_Float16)x0.z;  a[3]  = (_Float16)x0.w;
    a[4]  = (_Float16)x1.x;  a[5]  = (_Float16)x1.y;
    a[6]  = (_Float16)x1.z;  a[7]  = (_Float16)x1.w;
    a[8]  = (_Float16)x2.x;  a[9]  = (_Float16)x2.y;
    a[10] = (_Float16)x2.z;  a[11] = (_Float16)x2.w;
    a[12] = (_Float16)x3.x;  a[13] = (_Float16)x3.y;
    a[14] = (_Float16)x3.z;  a[15] = (_Float16)x3.w;
    return a;
}

// Stage W (192x128 f32, k-major) into LDS as f16 pre-swizzled into 48
// B-fragments (6 k-tiles x 8 col-tiles); fragment element i of lane l of frag
// f lives at ldsW[(f*32+l)*16+i].  B 32x16 layout: lanes 0-15 K=0..7,16..23,
// lanes 16-31 K=8..15,24..31, N = l%16.
__device__ __forceinline__ void stage_w(_Float16* ldsW, const float* __restrict__ Wg,
                                        int tid, int nthreads) {
    for (int idx = tid; idx < KDIM * DDIM; idx += nthreads) {
        int k    = idx >> 7;          // 0..191 (coalesced over col)
        int col  = idx & 127;
        int kt   = k >> 5;
        int kr   = k & 31;
        int lane = (((kr >> 3) & 1) << 4) | (col & 15);
        int i    = (kr & 7) | ((kr >> 4) << 3);
        int ct   = col >> 4;
        ldsW[((kt * 8 + ct) * 32 + lane) * 16 + i] = (_Float16)Wg[idx];
    }
}

// ---------------------------------------------------------------------------
// fc1:  h[n][a] = site[n]@fc1_w + fc1_b
// ---------------------------------------------------------------------------
__global__ void fc1_kernel(const float* __restrict__ site, const float* __restrict__ w,
                           const float* __restrict__ b, float* __restrict__ h) {
    int i = blockIdx.x * blockDim.x + threadIdx.x;
    if (i >= NN * AA) return;
    int n = i >> 6, a = i & 63;
    const float* row = site + (long)n * F0;
    float acc = b[a];
#pragma unroll 4
    for (int k = 0; k < F0; k++) acc += row[k] * w[k * AA + a];
    h[i] = acc;
}

// ---------------------------------------------------------------------------
// Pass 1: per-channel sum / sumsq of t = z@W + bw over all N*M rows.
// One wave computes a full 16x128 tile of t in registers (8 WMMA tiles x 6 k).
// ---------------------------------------------------------------------------
__global__ __launch_bounds__(256)
void conv_stats_kernel(const float* __restrict__ h, const float* __restrict__ bond,
                       const int* __restrict__ bidx, const float* __restrict__ Wg,
                       const float* __restrict__ bw,
                       float* __restrict__ gsum, float* __restrict__ gsq) {
    __shared__ _Float16 ldsW[48 * 512];
    __shared__ float bsum[DDIM], bsq[DDIM];
    int tid = threadIdx.x;
    stage_w(ldsW, Wg, tid, 256);
    if (tid < DDIM) { bsum[tid] = 0.f; bsq[tid] = 0.f; }
    __syncthreads();

    int wave = tid >> 5, lane = tid & 31;
    int r = (blockIdx.x * 8 + wave) * 16 + (lane & 15);
    int n = r / MM, m = r - n * MM;
    int j = bidx[n * MM + m];
    const float* pi = h + (long)n * AA;
    const float* pj = h + (long)j * AA;
    const float* pb = bond + (long)r * BB;
    int kb = (lane & 16) ? 8 : 0;

    v8f acc[8];
#pragma unroll
    for (int c = 0; c < 8; c++) acc[c] = {};

#pragma unroll
    for (int kt = 0; kt < 6; kt++) {
        const float* src = (kt < 2) ? pi : (kt < 4) ? pj : pb;
        v16h a = load_afrag(src + (kt & 1) * 32, kb);
#pragma unroll
        for (int ct = 0; ct < 8; ct++) {
            v16h b = *(const v16h*)&ldsW[((kt * 8 + ct) * 32 + lane) * 16];
            acc[ct] = wmma_f16(a, b, acc[ct]);
        }
    }

#pragma unroll
    for (int ct = 0; ct < 8; ct++) {
        int col = ct * 16 + (lane & 15);
        float bias = bw[col];
        float s = 0.f, q = 0.f;
#pragma unroll
        for (int v = 0; v < 8; v++) {
            float x = acc[ct][v] + bias;
            s += x; q += x * x;
        }
        atomicAdd(&bsum[col], s);
        atomicAdd(&bsq[col], q);
    }
    __syncthreads();
    if (tid < DDIM) { atomicAdd(&gsum[tid], bsum[tid]); atomicAdd(&gsq[tid], bsq[tid]); }
}

// ---------------------------------------------------------------------------
// BN finalize: scale = g*rsqrt(var+eps), shift = b - mean*scale
// ---------------------------------------------------------------------------
__global__ void bn_finalize_kernel(const float* __restrict__ sum, const float* __restrict__ sq,
                                   const float* __restrict__ g, const float* __restrict__ b,
                                   float* __restrict__ scale, float* __restrict__ shift,
                                   int nchan, float count) {
    int c = blockIdx.x * blockDim.x + threadIdx.x;
    if (c >= nchan) return;
    float m = sum[c] / count;
    float v = sq[c] / count - m * m;
    float sc = g[c] * rsqrtf(v + 1e-5f);
    scale[c] = sc;
    shift[c] = b[c] - m * sc;
}

// ---------------------------------------------------------------------------
// Pass 2: recompute t tiles, apply BN1, gate = sigmoid(filt)*softplus(core),
// sum over the 12 neighbors, emit s (N x 64) plus bn2 channel stats.
// Wave = (supertile of 4 sites = 48 rows = 3 WMMA row-tiles) x (col quarter q:
// filt cols q*16.. and core cols 64+q*16..).  Block = 8 waves = 2 supertiles.
// ---------------------------------------------------------------------------
__global__ __launch_bounds__(256)
void conv_apply_kernel(const float* __restrict__ h, const float* __restrict__ bond,
                       const int* __restrict__ bidx, const float* __restrict__ Wg,
                       const float* __restrict__ bw,
                       const float* __restrict__ scale1, const float* __restrict__ shift1,
                       float* __restrict__ s_out,
                       float* __restrict__ bn2sum, float* __restrict__ bn2sq) {
    extern __shared__ char smem[];
    _Float16* ldsW   = (_Float16*)smem;                       // 49152 B
    float*    gstage = (float*)(smem + 48 * 512 * 2);         // 8*48*16 f32 = 24576 B
    float*    lsum   = (float*)(smem + 48 * 512 * 2 + 24576); // 64
    float*    lsq    = lsum + 64;                             // 64

    int tid = threadIdx.x;
    stage_w(ldsW, Wg, tid, 256);
    if (tid < 64) { lsum[tid] = 0.f; lsq[tid] = 0.f; }
    __syncthreads();

    int wave = tid >> 5, lane = tid & 31;
    int st = blockIdx.x * 2 + (wave >> 2);   // supertile (4 sites)
    int q  = wave & 3;                       // column quarter
    int rowbase = st * 48;
    int kb = (lane & 16) ? 8 : 0;

    const float *pi[3], *pj[3], *pb[3];
#pragma unroll
    for (int tr = 0; tr < 3; tr++) {
        int r = rowbase + tr * 16 + (lane & 15);
        int n = r / MM, m = r - n * MM;
        int j = bidx[n * MM + m];
        pi[tr] = h + (long)n * AA;
        pj[tr] = h + (long)j * AA;
        pb[tr] = bond + (long)r * BB;
    }

    v8f accF[3], accC[3];
#pragma unroll
    for (int t = 0; t < 3; t++) { accF[t] = {}; accC[t] = {}; }

#pragma unroll
    for (int kt = 0; kt < 6; kt++) {
        v16h bF = *(const v16h*)&ldsW[((kt * 8 + q) * 32 + lane) * 16];
        v16h bC = *(const v16h*)&ldsW[((kt * 8 + q + 4) * 32 + lane) * 16];
        int off = (kt & 1) * 32;
#pragma unroll
        for (int tr = 0; tr < 3; tr++) {
            const float* src = (kt < 2) ? pi[tr] : (kt < 4) ? pj[tr] : pb[tr];
            v16h a = load_afrag(src + off, kb);
            accF[tr] = wmma_f16(a, bF, accF[tr]);
            accC[tr] = wmma_f16(a, bC, accC[tr]);
        }
    }

    // gate in registers, stage per-wave 48x16 into LDS
    int colF = q * 16 + (lane & 15);
    int colC = colF + AA;
    float bF_ = bw[colF], bC_ = bw[colC];
    float scF = scale1[colF], shF = shift1[colF];
    float scC = scale1[colC], shC = shift1[colC];
    float* gb = gstage + wave * 48 * 16;
    int hiofs = (lane & 16) ? 8 : 0;
#pragma unroll
    for (int tr = 0; tr < 3; tr++) {
#pragma unroll
        for (int v = 0; v < 8; v++) {
            float f = (accF[tr][v] + bF_) * scF + shF;
            float c = (accC[tr][v] + bC_) * scC + shC;
            gb[(tr * 16 + v + hiofs) * 16 + (lane & 15)] = sigmoidf(f) * softplusf(c);
        }
    }
    __syncthreads();

    // neighbor-sum: 64 tasks per wave = 4 sites x 16 cols, 2 per lane
#pragma unroll
    for (int t = 0; t < 2; t++) {
        int task = lane + t * 32;
        int nl = task >> 4, cc = task & 15;
        float s = 0.f;
#pragma unroll
        for (int m = 0; m < MM; m++) s += gb[(nl * MM + m) * 16 + cc];
        int a = q * 16 + cc;
        s_out[(long)(st * 4 + nl) * AA + a] = s;
        atomicAdd(&lsum[a], s);
        atomicAdd(&lsq[a], s * s);
    }
    __syncthreads();
    if (tid < 64) { atomicAdd(&bn2sum[tid], lsum[tid]); atomicAdd(&bn2sq[tid], lsq[tid]); }
}

// ---------------------------------------------------------------------------
// h = softplus(h + bn2(s))
// ---------------------------------------------------------------------------
__global__ void update_h_kernel(float* __restrict__ h, const float* __restrict__ s,
                                const float* __restrict__ scale2, const float* __restrict__ shift2) {
    int i = blockIdx.x * blockDim.x + threadIdx.x;
    if (i >= NN * AA) return;
    int a = i & 63;
    h[i] = softplusf(h[i] + s[i] * scale2[a] + shift2[a]);
}

// ---------------------------------------------------------------------------
// Pooling
// ---------------------------------------------------------------------------
__global__ void count_kernel(const int* __restrict__ cidx, int* __restrict__ cnt) {
    int n = blockIdx.x * blockDim.x + threadIdx.x;
    if (n < NN) atomicAdd(&cnt[cidx[n]], 1);
}

__global__ void pool_kernel(const float* __restrict__ h, const int* __restrict__ cidx,
                            float* __restrict__ pool) {
    int a = threadIdx.x;              // 64 threads = one channel each
    int n0 = blockIdx.x * 256;
    int n1 = n0 + 256; if (n1 > NN) n1 = NN;
    int cur = cidx[n0];
    float acc = 0.f;
    for (int n = n0; n < n1; n++) {
        int c = cidx[n];               // sorted -> few run breaks
        if (c != cur) { atomicAdd(&pool[cur * AA + a], acc); cur = c; acc = 0.f; }
        acc += h[(long)n * AA + a];
    }
    atomicAdd(&pool[cur * AA + a], acc);
}

// ---------------------------------------------------------------------------
// Head: feats = [pooled_mean | fap], top-BN over 128 rows, softplus, fc2,
// softplus, fc3.  Single block of 128 threads (one per crystal row).
// ---------------------------------------------------------------------------
__global__ __launch_bounds__(128)
void head_kernel(const float* __restrict__ pool, const int* __restrict__ cnt,
                 const float* __restrict__ fap,
                 const float* __restrict__ tg, const float* __restrict__ tb,
                 const float* __restrict__ fc2w, const float* __restrict__ fc2b,
                 const float* __restrict__ fc3w, const float* __restrict__ fc3b,
                 float* __restrict__ out) {
    __shared__ float feats[CC * (AA + 1)];
    __shared__ float sc[AA + 1], sh[AA + 1];
    int t = threadIdx.x;  // row 0..127
    float invc = 1.f / fmaxf((float)cnt[t], 1.f);
    for (int a = 0; a < AA; a++) feats[t * 65 + a] = pool[t * AA + a] * invc;
    feats[t * 65 + AA] = fap[t];
    __syncthreads();
    if (t < 65) {
        float s = 0.f, q = 0.f;
        for (int r = 0; r < CC; r++) { float x = feats[r * 65 + t]; s += x; q += x * x; }
        float m = s / (float)CC, v = q / (float)CC - m * m;
        float scl = tg[t] * rsqrtf(v + 1e-5f);
        sc[t] = scl; sh[t] = tb[t] - m * scl;
    }
    __syncthreads();
    for (int a = 0; a < 65; a++)
        feats[t * 65 + a] = softplusf(feats[t * 65 + a] * sc[a] + sh[a]);
    float pred = fc3b[0];
    for (int hc = 0; hc < HH; hc++) {
        float acc = fc2b[hc];
        for (int k = 0; k < 65; k++) acc += feats[t * 65 + k] * fc2w[k * HH + hc];
        pred += softplusf(acc) * fc3w[hc];
    }
    out[t] = pred;
}

// ---------------------------------------------------------------------------
// launch
// ---------------------------------------------------------------------------
extern "C" void kernel_launch(void* const* d_in, const int* in_sizes, int n_in,
                              void* d_out, int out_size, void* d_ws, size_t ws_size,
                              hipStream_t stream) {
    const float* site  = (const float*)d_in[0];
    const float* bond  = (const float*)d_in[1];
    const float* fap   = (const float*)d_in[2];
    const float* fc1w  = (const float*)d_in[3];
    const float* fc1b  = (const float*)d_in[4];
    const float* convw = (const float*)d_in[5];
    const float* convb = (const float*)d_in[6];
    const float* bn1g  = (const float*)d_in[7];
    const float* bn1b  = (const float*)d_in[8];
    const float* bn2g  = (const float*)d_in[9];
    const float* bn2b  = (const float*)d_in[10];
    const float* topg  = (const float*)d_in[11];
    const float* topb  = (const float*)d_in[12];
    const float* fc2w  = (const float*)d_in[13];
    const float* fc2b  = (const float*)d_in[14];
    const float* fc3w  = (const float*)d_in[15];
    const float* fc3b  = (const float*)d_in[16];
    const int*   bidx  = (const int*)d_in[17];
    const int*   cidx  = (const int*)d_in[18];
    float* out = (float*)d_out;

    float* wsf = (float*)d_ws;
    float* h      = wsf;                      // N*64
    float* s      = wsf + (long)NN * AA;      // N*64
    float* gsum   = s + (long)NN * AA;        // 128
    float* gsq    = gsum + 128;               // 128
    float* scale1 = gsq + 128;                // 128
    float* shift1 = scale1 + 128;             // 128
    float* b2sum  = shift1 + 128;             // 64
    float* b2sq   = b2sum + 64;               // 64
    float* scale2 = b2sq + 64;                // 64
    float* shift2 = scale2 + 64;              // 64
    float* pool   = shift2 + 64;              // 128*64
    int*   cnt    = (int*)(pool + CC * AA);   // 128

    // fc1
    fc1_kernel<<<(NN * AA + 255) / 256, 256, 0, stream>>>(site, fc1w, fc1b, h);

    size_t apply_smem = (size_t)48 * 512 * 2 + 24576 + 512;

    for (int l = 0; l < LL; l++) {
        const float* Wl  = convw + (long)l * KDIM * DDIM;
        const float* bwl = convb + (long)l * DDIM;
        hipMemsetAsync(gsum, 0, 2 * 128 * sizeof(float), stream);
        hipMemsetAsync(b2sum, 0, 2 * 64 * sizeof(float), stream);

        conv_stats_kernel<<<NM / 128, 256, 0, stream>>>(h, bond, bidx, Wl, bwl, gsum, gsq);
        bn_finalize_kernel<<<1, 128, 0, stream>>>(gsum, gsq, bn1g + l * DDIM, bn1b + l * DDIM,
                                                  scale1, shift1, DDIM, (float)NM);
        conv_apply_kernel<<<NN / 8, 256, apply_smem, stream>>>(h, bond, bidx, Wl, bwl,
                                                               scale1, shift1, s, b2sum, b2sq);
        bn_finalize_kernel<<<1, 64, 0, stream>>>(b2sum, b2sq, bn2g + l * AA, bn2b + l * AA,
                                                 scale2, shift2, AA, (float)NN);
        update_h_kernel<<<(NN * AA + 255) / 256, 256, 0, stream>>>(h, s, scale2, shift2);
    }

    hipMemsetAsync(pool, 0, (CC * AA) * sizeof(float) + CC * sizeof(int), stream);
    count_kernel<<<(NN + 255) / 256, 256, 0, stream>>>(cidx, cnt);
    pool_kernel<<<(NN + 255) / 256, 64, 0, stream>>>(h, cidx, pool);
    head_kernel<<<1, 128, 0, stream>>>(pool, cnt, fap, topg, topb, fc2w, fc2b, fc3w, fc3b, out);
}
